// SpatialGraphConv_4045859192983
// MI455X (gfx1250) — matrix operand order repaired and checked by reading, not compiled, not run on hardware
//
#include <hip/hip_runtime.h>

typedef __attribute__((ext_vector_type(16))) __bf16 v16bf;
typedef __attribute__((ext_vector_type(8)))  float  v8f;

// workspace layout (bytes)
#define WS_SUMS   0        // 128 f32: [0:64]=sum, [64:128]=sumsq
#define WS_SCALE  512      // 128 f32: [0:64]=scale, [64:128]=shift
#define WS_BIAS   1024     // 1600 f32: bias_eff[c*25+w]
#define WS_WT     7680     // 24 regions x 1024B : W as bf16 A-operand tiles
#define WS_AT     32256    // 6 regions x 1024B  : A as bf16 B-operand tiles

__device__ __forceinline__ unsigned short f2bf(float f) {
  unsigned int u = __builtin_bit_cast(unsigned int, f);
  u = (u + 0x7FFFu + ((u >> 16) & 1u)) >> 16;   // RNE
  return (unsigned short)u;
}

__device__ __forceinline__ v8f wmma_bf16(v16bf a, v16bf b, v8f c) {
  return __builtin_amdgcn_wmma_f32_16x16x32_bf16(false, a, false, b, (short)0, c,
                                                 false, false);
}

// ---------------------------------------------------------------------------
// Prep: swizzle W and A into WMMA operand layouts, fold bias through graph
// matmul (bias_eff[c][w] = sum_k b[k*64+c] * sum_v A[k,v,w]), zero stats.
// W A-operand tiles (k contiguous): lane = m + 16*(K>>4), half = K&15
// A B-operand tiles (k table order): lane = n + 16*(g&1), half = (K&7)+8*(g>>1), g=K>>3
// ---------------------------------------------------------------------------
__global__ __launch_bounds__(256) void prep_kernel(
    const float* __restrict__ W, const float* __restrict__ bvec,
    const float* __restrict__ A, unsigned char* __restrict__ ws) {
  const int i = blockIdx.x * 256 + threadIdx.x;        // 0..12287

  { // W (192x64) -> 24 A-operand tiles, one slot per thread
    const int region = i >> 9, sl = i & 511;
    const int ln = sl >> 4, hf = sl & 15;
    const int m  = ln & 15;
    const int K  = ((ln >> 4) << 4) | hf;
    const int o  = (region >> 1) * 16 + m;
    const int ci = (region & 1) * 32 + K;
    ((unsigned short*)(ws + WS_WT))[i] = f2bf(W[o * 64 + ci]);
  }
  if (i < 3072) { // A (3x25x25) -> 6 zero-padded B-operand tiles
    const int region = i >> 9, sl = i & 511;
    const int ln = sl >> 4, hf = sl & 15;
    const int k  = region >> 1, wt = region & 1;
    const int v  = (hf & 7) + ((ln >> 4) << 3) + ((hf >> 3) << 4);
    const int w  = wt * 16 + (ln & 15);
    const float val = (v < 25 && w < 25) ? A[(k * 25 + v) * 25 + w] : 0.f;
    ((unsigned short*)(ws + WS_AT))[i] = f2bf(val);
  }
  if (i < 1600) { // bias_eff
    const int c = i / 25, w = i % 25;
    float acc = 0.f;
    for (int k = 0; k < 3; ++k) {
      float cs = 0.f;
      for (int v = 0; v < 25; ++v) cs += A[(k * 25 + v) * 25 + w];
      acc += bvec[k * 64 + c] * cs;
    }
    ((float*)(ws + WS_BIAS))[i] = acc;
  }
  if (i < 128) ((float*)(ws + WS_SUMS))[i] = 0.f;
}

// ---------------------------------------------------------------------------
// Fused conv(1x1) + graph matmul + BN stats.
// Grid: (ceil(300/8)=38, B=64), 256 threads = 8 wave32.
// ---------------------------------------------------------------------------
__global__ __launch_bounds__(256, 1) void fused_kernel(
    const float* __restrict__ x, const unsigned char* __restrict__ ws,
    float* __restrict__ out, float* __restrict__ gsums) {
  __shared__ unsigned char lds[59904];
  unsigned short* Xs = (unsigned short*)lds;             // 2kk x 13pt x 1024B
  unsigned short* Ys = (unsigned short*)(lds + 26624);   // 32 regions x 1024B
  float* chS = (float*)(lds + 59392);
  float* chQ = (float*)(lds + 59648);

  const int tid  = threadIdx.x;
  const int wave = tid >> 5;
  const int lane = tid & 31;
  const int bi   = blockIdx.y;
  const int t0   = blockIdx.x * 8;
  const int ctile = wave & 3;            // this wave's 16-channel tile
  const int wtile = wave >> 2;           // this wave's 16-col w tile

  // zero Ys once: v>=25 pad slots must stay zero for all k iterations
  #pragma unroll
  for (int j = 0; j < 32; ++j) ((unsigned int*)Ys)[tid + 256 * j] = 0u;
  if (tid < 64) { chS[tid] = 0.f; chQ[tid] = 0.f; }

  // ---- stage X slab -> bf16 B-operand tiles in LDS ----
  // 32 channel-pairs x 52 pixel-quads: 2x global b128 + 4x packed ds b32 each
  const float* xb = x + bi * 64 * 7500;
  #pragma unroll
  for (int j = 0; j < 7; ++j) {
    const int e = tid + 256 * j;
    if (e < 1664) {
      const int cip = e / 52;            // channel pair 0..31
      const int p4  = e - cip * 52;      // pixel quad 0..51
      const int ci0 = cip * 2;
      const int p0  = p4 * 4;
      const int gp0 = t0 * 25 + p0;
      const float* r0 = xb + ci0 * 7500 + gp0;
      float a[4], b[4];
      if ((p0 + 3 < 200) && (gp0 + 3 < 7500)) {       // 16B-aligned fast path
        const float4 va = *reinterpret_cast<const float4*>(r0);
        const float4 vb = *reinterpret_cast<const float4*>(r0 + 7500);
        a[0] = va.x; a[1] = va.y; a[2] = va.z; a[3] = va.w;
        b[0] = vb.x; b[1] = vb.y; b[2] = vb.z; b[3] = vb.w;
      } else {
        #pragma unroll
        for (int q = 0; q < 4; ++q) {
          const bool ok = (p0 + q < 200) && (gp0 + q < 7500);
          a[q] = ok ? r0[q] : 0.f;
          b[q] = ok ? r0[7500 + q] : 0.f;
        }
      }
      const int kk  = ci0 >> 5;
      const int kc0 = ci0 & 31;          // even: pair shares one 16-half group
      const int pt  = p0 >> 4;
      const int lnb = (p0 & 15) + ((kc0 >> 4) << 4);
      unsigned int* dst = (unsigned int*)((unsigned char*)Xs +
                          (kk * 13 + pt) * 1024 + (kc0 & 15) * 2);
      #pragma unroll
      for (int q = 0; q < 4; ++q) {
        const unsigned int pk = (unsigned int)f2bf(a[q]) |
                                ((unsigned int)f2bf(b[q]) << 16);
        dst[(lnb + q) * 8] = pk;         // lane stride = 32B
      }
    }
  }
  __syncthreads();

  // preload graph B-operands (32B/lane each) and bias for this wave's tile
  v16bf bA[3];
  #pragma unroll
  for (int k = 0; k < 3; ++k)
    bA[k] = *reinterpret_cast<const v16bf*>(ws + WS_AT + (k * 2 + wtile) * 1024 + lane * 32);

  const int wcol = wtile * 16 + (lane & 15);
  const int mb   = (lane >> 4) << 3;              // D-layout row base
  float biasr[8];
  #pragma unroll
  for (int r = 0; r < 8; ++r) {
    const int c = ctile * 16 + mb + r;
    biasr[r] = (wcol < 25) ? ((const float*)(ws + WS_BIAS))[c * 25 + wcol] : 0.f;
  }

  const v8f zacc = {0.f, 0.f, 0.f, 0.f, 0.f, 0.f, 0.f, 0.f};
  v8f acc[8];
  #pragma unroll
  for (int t = 0; t < 8; ++t) acc[t] = zacc;

  #pragma unroll                          // full unroll: static bA[k]/W addressing
  for (int k = 0; k < 3; ++k) {
    // ---- GEMM1: Y_k = W_k (64x64) @ X (64x208), bf16 WMMA, f32 acc ----
    const int wot = (k * 4 + ctile) * 2;
    const v16bf wa0 = *reinterpret_cast<const v16bf*>(ws + WS_WT + (wot + 0) * 1024 + lane * 32);
    const v16bf wa1 = *reinterpret_cast<const v16bf*>(ws + WS_WT + (wot + 1) * 1024 + lane * 32);
    #pragma unroll
    for (int j = 0; j < 7; ++j) {
      const int pt = wtile + 2 * j;      // wave-uniform -> scalar branch,
      if (pt < 13) {                     // EXEC stays all-ones for WMMA
        const v16bf xb0 = *reinterpret_cast<const v16bf*>((unsigned char*)Xs + pt * 1024 + lane * 32);
        const v16bf xb1 = *reinterpret_cast<const v16bf*>((unsigned char*)Xs + (13 + pt) * 1024 + lane * 32);
        v8f y = zacc;
        y = wmma_bf16(wa0, xb0, y);
        y = wmma_bf16(wa1, xb1, y);
        // scatter D tile into Ys as GEMM2 A-operand: (m=c%16, K=v), per t
        const int pcol = pt * 16 + (lane & 15);
        if (pcol < 200) {
          const int tloc = pcol / 25;
          const int v    = pcol - tloc * 25;
          const int g1   = (v >> 3) & 1;
          const int hb   = (v & 7) + ((v >> 4) << 3);
          #pragma unroll
          for (int r = 0; r < 8; ++r) {
            const int ln = (mb + r) + (g1 << 4);
            Ys[(ctile * 8 + tloc) * 512 + ln * 16 + hb] = f2bf(y[r]);
          }
        }
      }
    }
    __syncthreads();
    // ---- GEMM2: acc[t] += Y_k[ctile,t] (16x32,v-pad) @ A_k[wtile] ----
    #pragma unroll
    for (int t = 0; t < 8; ++t) {
      const v16bf ya = *reinterpret_cast<const v16bf*>(
          (unsigned char*)Ys + (ctile * 8 + t) * 1024 + lane * 32);
      acc[t] = wmma_bf16(ya, bA[k], acc[t]);
    }
    __syncthreads();   // Ys reused next k
  }

  // ---- writeout + BN statistics ----
  float s[8], q[8];
  #pragma unroll
  for (int r = 0; r < 8; ++r) { s[r] = 0.f; q[r] = 0.f; }
  if (wcol < 25) {
    #pragma unroll
    for (int t = 0; t < 8; ++t) {
      const int tt = t0 + t;
      if (tt < 300) {
        #pragma unroll
        for (int r = 0; r < 8; ++r) {
          const int c = ctile * 16 + mb + r;
          const float val = acc[t][r] + biasr[r];
          out[((bi * 64 + c) * 300 + tt) * 25 + wcol] = val;
          s[r] += val; q[r] += val * val;
        }
      }
    }
    #pragma unroll
    for (int r = 0; r < 8; ++r) {
      const int c = ctile * 16 + mb + r;
      atomicAdd(&chS[c], s[r]);        // ds_add_f32
      atomicAdd(&chQ[c], q[r]);
    }
  }
  __syncthreads();
  if (tid < 64) {
    atomicAdd(&gsums[tid], chS[tid]);
    atomicAdd(&gsums[64 + tid], chQ[tid]);
  }
}

__global__ void finalize_kernel(const float* __restrict__ gsums,
                                const float* __restrict__ gamma,
                                const float* __restrict__ beta,
                                float* __restrict__ scaleshift) {
  const int c = threadIdx.x;                        // 64 threads
  const float n    = 64.f * 300.f * 25.f;
  const float mean = gsums[c] / n;
  const float var  = gsums[64 + c] / n - mean * mean;
  const float sc   = gamma[c] * rsqrtf(var + 1e-5f);
  scaleshift[c]      = sc;
  scaleshift[64 + c] = beta[c] - mean * sc;
}

__global__ __launch_bounds__(256) void apply_kernel(
    float* __restrict__ out, const float* __restrict__ scaleshift) {
  __shared__ float sc[64], sh[64];
  if (threadIdx.x < 64) {
    sc[threadIdx.x] = scaleshift[threadIdx.x];
    sh[threadIdx.x] = scaleshift[64 + threadIdx.x];
  }
  __syncthreads();
  const int base = (blockIdx.x * 256 + threadIdx.x) * 4;   // < 30,720,000
  float4 v = *reinterpret_cast<float4*>(out + base);
  float* pv = &v.x;
  #pragma unroll
  for (int j = 0; j < 4; ++j) {
    const int c = ((base + j) / 7500) & 63;                // magic-mul divide
    pv[j] = pv[j] * sc[c] + sh[c];
  }
  *reinterpret_cast<float4*>(out + base) = v;
}

extern "C" void kernel_launch(void* const* d_in, const int* in_sizes, int n_in,
                              void* d_out, int out_size, void* d_ws, size_t ws_size,
                              hipStream_t stream) {
  const float* x     = (const float*)d_in[0];
  const float* W     = (const float*)d_in[1];
  const float* bvec  = (const float*)d_in[2];
  const float* A     = (const float*)d_in[3];
  const float* gamma = (const float*)d_in[4];
  const float* beta  = (const float*)d_in[5];
  float* out = (float*)d_out;
  unsigned char* ws = (unsigned char*)d_ws;

  prep_kernel<<<48, 256, 0, stream>>>(W, bvec, A, ws);
  fused_kernel<<<dim3(38, 64), 256, 0, stream>>>(x, ws, out, (float*)(ws + WS_SUMS));
  finalize_kernel<<<1, 64, 0, stream>>>((const float*)(ws + WS_SUMS), gamma, beta,
                                        (float*)(ws + WS_SCALE));
  apply_kernel<<<30000, 256, 0, stream>>>(out, (const float*)(ws + WS_SCALE));
}
